// Attention_36464272343044
// MI455X (gfx1250) — matrix operand order repaired and checked by reading, not compiled, not run on hardware
//
#include <hip/hip_runtime.h>
#include <hip/hip_bf16.h>

typedef __attribute__((ext_vector_type(16))) _Float16 v16h;
typedef __attribute__((ext_vector_type(8)))  _Float16 v8h;
typedef __attribute__((ext_vector_type(8)))  float    v8f;

#define N_TOK 4096
#define C_DIM 256
#define H_NUM 8
#define DH    32
#define MASK_NEG -9e15f
#define SCALE 0.17677669529663687f   // 32^-0.5

// ---------- WMMA helpers (contiguous k = 16*(lane>>4)+s for A and B, not the ISA table order) ----------

__device__ __forceinline__ v8f wmma_f16(v16h a, v16h b, v8f c) {
    return __builtin_amdgcn_wmma_f32_16x16x32_f16(false, a, false, b, (short)0, c, false, false);
}

// A-matrix 16x32 f16: lane holds row m=lane&15; K = kbase+[0..15],
// kbase = (lane>>4)*16. `base` must already point at row + kbase.
__device__ __forceinline__ v16h load_a16(const _Float16* base) {
    v8h lo = *(const v8h*)base;
    v8h hb = *(const v8h*)(base + 8);
    v16h r;
#pragma unroll
    for (int i = 0; i < 8; ++i) { r[i] = lo[i]; r[i + 8] = hb[i]; }
    return r;
}

// B-matrix 32x16 f16: lane holds column n=lane&15; K = (lane>>4)*16 + [0..15],
// 16 contiguous halfs at `base`.
__device__ __forceinline__ v16h load_b16(const _Float16* base) {
    v8h lo = *(const v8h*)base;
    v8h hb = *(const v8h*)(base + 8);
    v16h r;
#pragma unroll
    for (int i = 0; i < 8; ++i) { r[i] = lo[i]; r[i + 8] = hb[i]; }
    return r;
}

__device__ __forceinline__ float redmax16(float v) {
#pragma unroll
    for (int m = 1; m < 16; m <<= 1) v = fmaxf(v, __shfl_xor(v, m, 32));
    return v;
}

// ---------------------------- f32 -> f16 convert ----------------------------

__global__ void cvt_f16(const float* __restrict__ in, _Float16* __restrict__ out, int n) {
    int i = blockIdx.x * blockDim.x + threadIdx.x;
    if (i < n) out[i] = (_Float16)in[i];
}

// ------------------------------- QKV GEMM ----------------------------------
// qkv[n,d] = sum_c x[n,c] * w_qkv[d,c]; d = sec*256 + h*32 + dh.
// Scatter into q[h][n][32], k[h][n][32], vT[h][32][n] (all f16).

__global__ __launch_bounds__(256) void qkv_gemm(
    const _Float16* __restrict__ xh, const _Float16* __restrict__ wh,
    _Float16* __restrict__ qb, _Float16* __restrict__ kb, _Float16* __restrict__ vT)
{
    const int lane = threadIdx.x & 31, wave = threadIdx.x >> 5;
    const int tile = blockIdx.x * 8 + wave;         // 0..12287
    const int n0 = (tile & 255) * 16;
    const int d0 = (tile >> 8) * 16;                // 0..47 tiles of d
    const int col = lane & 15, hi = lane >> 4;

    v8f acc = {};
#pragma unroll
    for (int kc = 0; kc < C_DIM; kc += 32) {
        v16h a = load_a16(xh + (size_t)(n0 + col) * C_DIM + kc + hi * 16);
        v16h b = load_b16(wh + (size_t)(d0 + col) * C_DIM + kc + hi * 16);
        acc = wmma_f16(a, b, acc);
    }
    const int d   = d0 + col;
    const int sec = d >> 8, cc = d & 255;
    const int h = cc >> 5, dh = cc & 31;
#pragma unroll
    for (int r = 0; r < 8; ++r) {
        const int n = n0 + r + 8 * hi;
        const _Float16 val = (_Float16)acc[r];
        if (sec == 0)      qb[((size_t)h * N_TOK + n) * DH + dh] = val;
        else if (sec == 1) kb[((size_t)h * N_TOK + n) * DH + dh] = val;
        else               vT[((size_t)h * DH + dh) * N_TOK + n] = val;
    }
}

// --------------------------- fused flash attention --------------------------
// One wave = one (head, 16-row query block). Keys processed 64 at a time:
// 4 score WMMAs + online softmax + 2 row-sum WMMAs (P x ones) + 4 PV WMMAs.
// The adj mask (64 MB, dominant HBM stream) is double-buffered into LDS with
// GLOBAL_LOAD_ASYNC_TO_LDS_B128 (ASYNCcnt), one chunk ahead of compute.

// Issue one 16x64-int adj tile (4 KB) as 8 wave-wide async b128 copies.
__device__ __forceinline__ void async_adj_tile(const int* __restrict__ adj,
                                               int n0, int kk,
                                               int* __restrict__ ldsbuf, int lane)
{
#pragma unroll
    for (int j = 0; j < 8; ++j) {
        const int idx = j * 32 + lane;        // 16-byte chunk id (0..255)
        const int r   = idx >> 4;             // tile row 0..15
        const int c4  = (idx & 15) * 4;       // int column 0,4,..,60
        const unsigned ldsa =
            (unsigned)(uintptr_t)(ldsbuf + r * 64 + c4);          // LDS byte offset
        const unsigned long long ga =
            (unsigned long long)(uintptr_t)(adj + (size_t)(n0 + r) * N_TOK + kk + c4);
        asm volatile("global_load_async_to_lds_b128 %0, %1, off"
                     :: "v"(ldsa), "v"(ga) : "memory");
    }
}

__global__ __launch_bounds__(256) void attn_kernel(
    const _Float16* __restrict__ q, const _Float16* __restrict__ k,
    const _Float16* __restrict__ vT, const int* __restrict__ adj,
    _Float16* __restrict__ ao)
{
    __shared__ int      adjT[8][2][16 * 64];   // per-wave double-buffered mask tile (64KB)
    __shared__ _Float16 pstage[8][16 * 64];    // per-wave P staging tile (16KB)

    const int lane = threadIdx.x & 31, wave = threadIdx.x >> 5;
    const int h  = blockIdx.y;
    const int n0 = (blockIdx.x * 8 + wave) * 16;
    const int col = lane & 15, hi = lane >> 4;

    const _Float16* qh = q  + (size_t)h * N_TOK * DH;
    const _Float16* kh = k  + (size_t)h * N_TOK * DH;
    const _Float16* vh = vT + (size_t)h * DH * N_TOK;

    const v16h aq = load_a16(qh + (size_t)(n0 + col) * DH + hi * 16);

    v16h bones;                                 // all-ones B matrix for row sums
#pragma unroll
    for (int i = 0; i < 16; ++i) bones[i] = (_Float16)1.0f;

    v8f o_lo = {}, o_hi = {}, lt = {};          // lt: row-sum accumulator (C layout)
    float m8[8];
#pragma unroll
    for (int r = 0; r < 8; ++r) m8[r] = -INFINITY;

    _Float16* L = pstage[wave];

    // prime the async pipeline with the first mask tile
    async_adj_tile(adj, n0, 0, adjT[wave][0], lane);

    for (int kk = 0; kk < N_TOK; kk += 64) {
        const int cur = (kk >> 6) & 1;

        // prefetch next mask tile while we compute this chunk
        if (kk + 64 < N_TOK)
            async_adj_tile(adj, n0, kk + 64, adjT[wave][cur ^ 1], lane);

        // ---- scores S = Q * K^T for key columns kk..kk+63 (overlaps async) ----
        v8f s[4];
#pragma unroll
        for (int c = 0; c < 4; ++c) {
            v16h bk = load_b16(kh + (size_t)(kk + c * 16 + col) * DH + hi * 16);
            v8f z = {};
            s[c] = wmma_f16(aq, bk, z);
        }

        // current tile's 8 async copies are the oldest outstanding ops
        if (kk + 64 < N_TOK)
            asm volatile("s_wait_asynccnt 0x8" ::: "memory");
        else
            asm volatile("s_wait_asynccnt 0x0" ::: "memory");
        __builtin_amdgcn_wave_barrier();

        // ---- mask + scale + online max (rows r+8*hi live in this lane half) ----
        const int* arow = adjT[wave][cur];
#pragma unroll
        for (int r = 0; r < 8; ++r) {
            const int* ap = arow + (r + 8 * hi) * 64 + col;
            const float t0 = (ap[0]  > 0) ? s[0][r] * SCALE : MASK_NEG;
            const float t1 = (ap[16] > 0) ? s[1][r] * SCALE : MASK_NEG;
            const float t2 = (ap[32] > 0) ? s[2][r] * SCALE : MASK_NEG;
            const float t3 = (ap[48] > 0) ? s[3][r] * SCALE : MASK_NEG;
            const float tmax = redmax16(fmaxf(fmaxf(t0, t1), fmaxf(t2, t3)));
            const float mnew = fmaxf(m8[r], tmax);
            const float alpha = __expf(m8[r] - mnew);
            m8[r] = mnew;
            s[0][r] = __expf(t0 - mnew);
            s[1][r] = __expf(t1 - mnew);
            s[2][r] = __expf(t2 - mnew);
            s[3][r] = __expf(t3 - mnew);
            o_lo[r] *= alpha; o_hi[r] *= alpha; lt[r] *= alpha;
        }

        // ---- C-layout -> A-layout for P (16x64) via LDS bounce ----
#pragma unroll
        for (int r = 0; r < 8; ++r) {
            _Float16* Lr = L + (r + 8 * hi) * 64 + col;
            Lr[0]  = (_Float16)s[0][r];
            Lr[16] = (_Float16)s[1][r];
            Lr[32] = (_Float16)s[2][r];
            Lr[48] = (_Float16)s[3][r];
        }
        asm volatile("s_wait_dscnt 0" ::: "memory");
        __builtin_amdgcn_wave_barrier();
        v16h pa0 = load_a16(L + col * 64 + hi * 16);        // keys kk..kk+31
        v16h pa1 = load_a16(L + col * 64 + 32 + hi * 16);   // keys kk+32..kk+63
        __builtin_amdgcn_wave_barrier();
        asm volatile("s_wait_dscnt 0" ::: "memory");

        // ---- row sums via WMMA with all-ones B (replaces shuffle-tree sums) ----
        lt = wmma_f16(pa0, bones, lt);
        lt = wmma_f16(pa1, bones, lt);

        // ---- O += P * V  (V^T rows are contiguous B columns) ----
        v16h bv;
        bv = load_b16(vh + (size_t)col * N_TOK + kk + hi * 16);
        o_lo = wmma_f16(pa0, bv, o_lo);
        bv = load_b16(vh + (size_t)col * N_TOK + kk + 32 + hi * 16);
        o_lo = wmma_f16(pa1, bv, o_lo);
        bv = load_b16(vh + (size_t)(col + 16) * N_TOK + kk + hi * 16);
        o_hi = wmma_f16(pa0, bv, o_hi);
        bv = load_b16(vh + (size_t)(col + 16) * N_TOK + kk + 32 + hi * 16);
        o_hi = wmma_f16(pa1, bv, o_hi);
    }

    // ---- normalize and store [n][h*32 + dh] as f16 for the proj GEMM ----
#pragma unroll
    for (int r = 0; r < 8; ++r) {
        const float inv = 1.0f / lt[r];
        const size_t row = (size_t)(n0 + r + 8 * hi) * C_DIM + h * DH;
        ao[row + col]      = (_Float16)(o_lo[r] * inv);
        ao[row + 16 + col] = (_Float16)(o_hi[r] * inv);
    }
}

// ------------------------------- output proj --------------------------------

__global__ __launch_bounds__(256) void proj_gemm(
    const _Float16* __restrict__ ao, const _Float16* __restrict__ wh,
    const float* __restrict__ bias, float* __restrict__ out)
{
    const int lane = threadIdx.x & 31, wave = threadIdx.x >> 5;
    const int tile = blockIdx.x * 8 + wave;         // 0..4095
    const int n0 = (tile & 255) * 16;
    const int d0 = (tile >> 8) * 16;                // 0..15 tiles of d
    const int col = lane & 15, hi = lane >> 4;

    v8f acc = {};
#pragma unroll
    for (int kc = 0; kc < C_DIM; kc += 32) {
        v16h a = load_a16(ao + (size_t)(n0 + col) * C_DIM + kc + hi * 16);
        v16h b = load_b16(wh + (size_t)(d0 + col) * C_DIM + kc + hi * 16);
        acc = wmma_f16(a, b, acc);
    }
    const float bv = bias[d0 + col];
#pragma unroll
    for (int r = 0; r < 8; ++r)
        out[(size_t)(n0 + r + 8 * hi) * C_DIM + d0 + col] = acc[r] + bv;
}

// --------------------------------- launch -----------------------------------

extern "C" void kernel_launch(void* const* d_in, const int* in_sizes, int n_in,
                              void* d_out, int out_size, void* d_ws, size_t ws_size,
                              hipStream_t stream) {
    const float* x      = (const float*)d_in[0];
    const int*   adj    = (const int*)d_in[1];
    const float* w_qkv  = (const float*)d_in[2];
    const float* w_proj = (const float*)d_in[3];
    const float* b_proj = (const float*)d_in[4];
    float* out = (float*)d_out;

    char* ws = (char*)d_ws;
    _Float16* x16    = (_Float16*)(ws);                       // 4096*256*2   = 2 MB
    _Float16* wqkv16 = (_Float16*)(ws + (2u << 20));          // 768*256*2    = 384 KB
    _Float16* wprj16 = (_Float16*)(ws + (2u << 20) + 393216); // 256*256*2    = 128 KB
    _Float16* qb     = (_Float16*)(ws + (2u << 20) + 524288); // 2 MB
    _Float16* kb     = (_Float16*)(ws + (4u << 20) + 524288); // 2 MB
    _Float16* vT     = (_Float16*)(ws + (6u << 20) + 524288); // 2 MB
    _Float16* ao     = (_Float16*)(ws + (8u << 20) + 524288); // 2 MB (total ~10.5 MB)

    cvt_f16<<<(N_TOK * C_DIM + 255) / 256, 256, 0, stream>>>(x, x16, N_TOK * C_DIM);
    cvt_f16<<<(3 * C_DIM * C_DIM + 255) / 256, 256, 0, stream>>>(w_qkv, wqkv16, 3 * C_DIM * C_DIM);
    cvt_f16<<<(C_DIM * C_DIM + 255) / 256, 256, 0, stream>>>(w_proj, wprj16, C_DIM * C_DIM);

    // (N/16)*(768/16) = 12288 wave-tiles, 8 waves/block
    qkv_gemm<<<1536, 256, 0, stream>>>(x16, wqkv16, qb, kb, vT);

    // 8 heads x 256 query blocks, 8 waves/block
    attn_kernel<<<dim3(32, H_NUM), 256, 0, stream>>>(qb, kb, vT, adj, ao);

    // (N/16)*(256/16) = 4096 wave-tiles
    proj_gemm<<<512, 256, 0, stream>>>(ao, wprj16, b_proj, out);
}
